// MPNN_10651518894614
// MI455X (gfx1250) — matrix operand order repaired and checked by reading, not compiled, not run on hardware
//
#include <hip/hip_runtime.h>
#include <hip/hip_bf16.h>
#include <math.h>

// Problem constants (from reference)
#define BB 16
#define MM 256
#define NN 512
#define H1 16
#define H2 8
#define UU 8
#define OUTC 4
#define LL 2

typedef float v2f  __attribute__((ext_vector_type(2)));
typedef float v8f  __attribute__((ext_vector_type(8)));
typedef __bf16 v16bf __attribute__((ext_vector_type(16)));

// ---------------------------------------------------------------------------
// Kernel 1: G = A^T A per batch, via WMMA (f32 in / f32 accum).
// One wave32 block computes a 32x32 block of G[b] as a 2x2 grid of 16x16
// WMMA tiles: per K-step of 4 we load 2 A-operands + 2 B-operands (8 VMEM)
// and issue 4 v_wmma_f32_16x16x4_f32 into 4 independent accumulators
// (2 loads/WMMA, no serial WMMA dependency chain).
// VGPR layouts (ISA 7.12.2): A-op 16x4 = 2 VGPRs, lanes 0-15 K=0,1 /
// lanes 16-31 K=2,3, M=lane%16. B-op 4x16 = 2 VGPRs, v0 rows K=0|2,
// v1 rows K=1|3, N=lane%16. C/D 16x16 = 8 VGPRs, VGPR v = row v (lo lanes)
// and v+8 (hi lanes).
// ---------------------------------------------------------------------------
__global__ __launch_bounds__(32) void gram_kernel(const float* __restrict__ A,
                                                  float* __restrict__ G)
{
    const int jt = blockIdx.x * 32;          // output column block (j)
    const int it = blockIdx.y * 32;          // output row block (i)
    const int b  = blockIdx.z;
    const int lane = threadIdx.x;
    const int lo = lane & 15;
    const int hi = lane >> 4;

    const float* __restrict__ Ab = A + (size_t)b * MM * NN;

    v8f acc00 = {0.f,0.f,0.f,0.f,0.f,0.f,0.f,0.f};
    v8f acc01 = acc00, acc10 = acc00, acc11 = acc00;

#if __has_builtin(__builtin_amdgcn_wmma_f32_16x16x4_f32)
    #pragma unroll 4
    for (int m0 = 0; m0 < MM; m0 += 4) {
        const float* __restrict__ row0 = Ab + (size_t)(m0 + 2 * hi) * NN;
        const float* __restrict__ row1 = row0 + NN;
        v2f a0, a1, b0, b1;
        // A-operands: columns of A (A^T rows) for the two i-tiles
        a0.x = row0[it + lo];       a0.y = row1[it + lo];
        a1.x = row0[it + 16 + lo];  a1.y = row1[it + 16 + lo];
        // B-operands: rows of A for the two j-tiles
        b0.x = row0[jt + lo];       b0.y = row1[jt + lo];
        b1.x = row0[jt + 16 + lo];  b1.y = row1[jt + 16 + lo];
        acc00 = __builtin_amdgcn_wmma_f32_16x16x4_f32(false, a0, false, b0,
                                                      (short)0, acc00, false, false);
        acc01 = __builtin_amdgcn_wmma_f32_16x16x4_f32(false, a0, false, b1,
                                                      (short)0, acc01, false, false);
        acc10 = __builtin_amdgcn_wmma_f32_16x16x4_f32(false, a1, false, b0,
                                                      (short)0, acc10, false, false);
        acc11 = __builtin_amdgcn_wmma_f32_16x16x4_f32(false, a1, false, b1,
                                                      (short)0, acc11, false, false);
    }
#else
    // Fallback: bf16 inputs, f32 accumulation, K-step 32.
    #pragma unroll 2
    for (int m0 = 0; m0 < MM; m0 += 32) {
        v16bf a0, a1, b0, b1;
        #pragma unroll
        for (int e = 0; e < 16; ++e) {
            const int ka = (e < 8) ? (8 * hi + e) : (16 + 8 * hi + (e - 8));
            const int kb = ka;
            a0[e] = (__bf16)Ab[(size_t)(m0 + ka) * NN + it + lo];
            a1[e] = (__bf16)Ab[(size_t)(m0 + ka) * NN + it + 16 + lo];
            b0[e] = (__bf16)Ab[(size_t)(m0 + kb) * NN + jt + lo];
            b1[e] = (__bf16)Ab[(size_t)(m0 + kb) * NN + jt + 16 + lo];
        }
        acc00 = __builtin_amdgcn_wmma_f32_16x16x32_bf16(false, a0, false, b0,
                                                        (short)0, acc00, false, false);
        acc01 = __builtin_amdgcn_wmma_f32_16x16x32_bf16(false, a0, false, b1,
                                                        (short)0, acc01, false, false);
        acc10 = __builtin_amdgcn_wmma_f32_16x16x32_bf16(false, a1, false, b0,
                                                        (short)0, acc10, false, false);
        acc11 = __builtin_amdgcn_wmma_f32_16x16x32_bf16(false, a1, false, b1,
                                                        (short)0, acc11, false, false);
    }
#endif

    float* __restrict__ Gb = G + ((size_t)b * NN + it) * NN + jt;
    #pragma unroll
    for (int v = 0; v < 8; ++v) {
        const int row = v + 8 * hi;                      // M index within tile
        Gb[(size_t)row * NN + lo]               = acc00[v];
        Gb[(size_t)row * NN + 16 + lo]          = acc01[v];
        Gb[(size_t)(row + 16) * NN + lo]        = acc10[v];
        Gb[(size_t)(row + 16) * NN + 16 + lo]   = acc11[v];
    }
}

// ---------------------------------------------------------------------------
// Kernel 2: ATy, diag(G), sigma2 -> u0 = init_feat @ w_init^T + b_init
// One thread per (b, n) row.
// ---------------------------------------------------------------------------
__global__ void init_kernel(const float* __restrict__ A,
                            const float* __restrict__ y,
                            const float* __restrict__ sigma2,
                            const float* __restrict__ G,
                            const float* __restrict__ w_init,
                            const float* __restrict__ b_init,
                            float* __restrict__ u)
{
    const int p = blockIdx.x * blockDim.x + threadIdx.x;   // b*NN + n
    if (p >= BB * NN) return;
    const int b = p >> 9;
    const int n = p & (NN - 1);

    float aty = 0.f;
    const float* __restrict__ Ab = A + (size_t)b * MM * NN + n;
    const float* __restrict__ yb = y + (size_t)b * MM;
    for (int m = 0; m < MM; ++m)
        aty = fmaf(Ab[(size_t)m * NN], yb[m], aty);

    const float dg = G[((size_t)b * NN + n) * NN + n];
    const float s  = sigma2[b];

    #pragma unroll
    for (int o = 0; o < UU; ++o) {
        const float* w = w_init + o * 3;
        u[(size_t)p * UU + o] = fmaf(aty, w[0], fmaf(dg, w[1], fmaf(s, w[2], b_init[o])));
    }
}

// ---------------------------------------------------------------------------
// Kernel 3: factor pair-MLP layer 1.
// Pn[b,n,:] = b_p1 + sigma2[b]*w_p1[:,17] + u[b,n]@w_p1[:,0:8]^T   (bias folded)
// Pj[b,n,:] =                               u[b,n]@w_p1[:,8:16]^T
// ---------------------------------------------------------------------------
__global__ void proj_kernel(const float* __restrict__ u,
                            const float* __restrict__ sigma2,
                            const float* __restrict__ w_p1,
                            const float* __restrict__ b_p1,
                            float* __restrict__ Pn,
                            float* __restrict__ Pj)
{
    const int p = blockIdx.x * blockDim.x + threadIdx.x;
    if (p >= BB * NN) return;
    const int b = p >> 9;

    float uv[UU];
    #pragma unroll
    for (int k = 0; k < UU; ++k) uv[k] = u[(size_t)p * UU + k];
    const float s = sigma2[b];

    #pragma unroll
    for (int h = 0; h < H1; ++h) {
        const float* w = w_p1 + h * (2 * UU + 2);
        float pn = fmaf(s, w[17], b_p1[h]);
        float pj = 0.f;
        #pragma unroll
        for (int k = 0; k < UU; ++k) {
            pn = fmaf(uv[k], w[k], pn);
            pj = fmaf(uv[k], w[UU + k], pj);
        }
        Pn[(size_t)p * H1 + h] = pn;
        Pj[(size_t)p * H1 + h] = pj;
    }
}

// ---------------------------------------------------------------------------
// Kernel 4: per-pair MLP (layers 1-3) + sum over j (off-diagonal only).
// One block per (b, i); 256 threads sweep j in two strides of 256.
// Weights + Pn[b,i] staged in LDS; 8-wide message reduced via LDS tree.
// ---------------------------------------------------------------------------
__global__ __launch_bounds__(256) void pair_msg_kernel(
    const float* __restrict__ G,  const float* __restrict__ Pn,
    const float* __restrict__ Pj, const float* __restrict__ w_p1,
    const float* __restrict__ w_p2, const float* __restrict__ b_p2,
    const float* __restrict__ w_p3, const float* __restrict__ b_p3,
    float* __restrict__ m_sum)
{
    const int bi = blockIdx.x;           // b*NN + i
    const int b  = bi >> 9;
    const int i  = bi & (NN - 1);
    const int tid = threadIdx.x;

    __shared__ float sPn[H1], sWg[H1];
    __shared__ float sW2[H2 * H1], sB2[H2];
    __shared__ float sW3[UU * H2], sB3[UU];
    __shared__ float red[256 * UU];

    if (tid < H1) {
        sPn[tid] = Pn[(size_t)bi * H1 + tid];
        sWg[tid] = w_p1[tid * (2 * UU + 2) + 16];      // G-feature column
    }
    if (tid >= 32  && tid < 32 + H2 * H1) sW2[tid - 32]  = w_p2[tid - 32];
    if (tid >= 160 && tid < 160 + H2)     sB2[tid - 160] = b_p2[tid - 160];
    if (tid >= 168 && tid < 168 + UU*H2)  sW3[tid - 168] = w_p3[tid - 168];
    if (tid >= 232 && tid < 232 + UU)     sB3[tid - 232] = b_p3[tid - 232];
    __syncthreads();

    float acc[UU];
    #pragma unroll
    for (int o = 0; o < UU; ++o) acc[o] = 0.f;

    const float* __restrict__ Gi = G + ((size_t)b * NN + i) * NN;

    #pragma unroll
    for (int jj = 0; jj < NN / 256; ++jj) {
        const int j = tid + jj * 256;
        if (j == i) continue;                          // off_diag mask
        const float gij = Gi[j];
        const float* __restrict__ pj = Pj + ((size_t)b * NN + j) * H1;

        float h1[H1];
        #pragma unroll
        for (int h = 0; h < H1; ++h) {
            const float v = fmaf(gij, sWg[h], sPn[h] + pj[h]);
            h1[h] = v > 0.f ? v : 0.f;
        }
        float h2[H2];
        #pragma unroll
        for (int o = 0; o < H2; ++o) {
            float t = sB2[o];
            #pragma unroll
            for (int h = 0; h < H1; ++h) t = fmaf(h1[h], sW2[o * H1 + h], t);
            h2[o] = t > 0.f ? t : 0.f;
        }
        #pragma unroll
        for (int o = 0; o < UU; ++o) {
            float t = sB3[o];
            #pragma unroll
            for (int k = 0; k < H2; ++k) t = fmaf(h2[k], sW3[o * H2 + k], t);
            acc[o] += t;
        }
    }

    #pragma unroll
    for (int o = 0; o < UU; ++o) red[tid * UU + o] = acc[o];
    __syncthreads();
    for (int s = 128; s > 0; s >>= 1) {
        if (tid < s) {
            #pragma unroll
            for (int o = 0; o < UU; ++o)
                red[tid * UU + o] += red[(tid + s) * UU + o];
        }
        __syncthreads();
    }
    if (tid < UU) m_sum[(size_t)bi * UU + tid] = red[tid];
}

// ---------------------------------------------------------------------------
// Kernel 5: GRU cell update + u = g @ w_up^T + b_up. One thread per row.
// first==1 -> hidden state starts at zero (never trusts poisoned workspace).
// ---------------------------------------------------------------------------
__global__ void gru_kernel(const float* __restrict__ m_sum,
                           const float* __restrict__ r,
                           const float* __restrict__ Sig,
                           const float* __restrict__ w_ih, const float* __restrict__ b_ih,
                           const float* __restrict__ w_hh, const float* __restrict__ b_hh,
                           const float* __restrict__ w_up, const float* __restrict__ b_up,
                           float* __restrict__ g, float* __restrict__ u, int first)
{
    const int p = blockIdx.x * blockDim.x + threadIdx.x;
    if (p >= BB * NN) return;

    float in[UU + 2];
    #pragma unroll
    for (int k = 0; k < UU; ++k) in[k] = m_sum[(size_t)p * UU + k];
    in[UU]     = r[p];
    in[UU + 1] = Sig[p];

    float gp[H1];
    #pragma unroll
    for (int h = 0; h < H1; ++h) gp[h] = first ? 0.f : g[(size_t)p * H1 + h];

    float gi[3 * H1], gh[3 * H1];
    #pragma unroll
    for (int row = 0; row < 3 * H1; ++row) {
        float ti = b_ih[row];
        const float* wi = w_ih + row * (UU + 2);
        #pragma unroll
        for (int k = 0; k < UU + 2; ++k) ti = fmaf(in[k], wi[k], ti);
        gi[row] = ti;
        float th = b_hh[row];
        const float* wh = w_hh + row * H1;
        #pragma unroll
        for (int h = 0; h < H1; ++h) th = fmaf(gp[h], wh[h], th);
        gh[row] = th;
    }

    #pragma unroll
    for (int h = 0; h < H1; ++h) {
        const float rr = 1.f / (1.f + __expf(-(gi[h] + gh[h])));
        const float zz = 1.f / (1.f + __expf(-(gi[H1 + h] + gh[H1 + h])));
        const float nn = tanhf(fmaf(rr, gh[2 * H1 + h], gi[2 * H1 + h]));
        const float gn = fmaf(zz, gp[h] - nn, nn);     // (1-zz)*nn + zz*gp
        g[(size_t)p * H1 + h] = gn;
        gp[h] = gn;
    }

    #pragma unroll
    for (int o = 0; o < UU; ++o) {
        float t = b_up[o];
        const float* w = w_up + o * H1;
        #pragma unroll
        for (int h = 0; h < H1; ++h) t = fmaf(gp[h], w[h], t);
        u[(size_t)p * UU + o] = t;
    }
}

// ---------------------------------------------------------------------------
// Kernel 6: readout MLP + softmax; also emits u and g into concatenated out.
// ---------------------------------------------------------------------------
__global__ void readout_kernel(const float* __restrict__ u,
                               const float* __restrict__ g,
                               const float* __restrict__ w_r1, const float* __restrict__ b_r1,
                               const float* __restrict__ w_r2, const float* __restrict__ b_r2,
                               const float* __restrict__ w_r3, const float* __restrict__ b_r3,
                               float* __restrict__ out_probs,
                               float* __restrict__ out_u,
                               float* __restrict__ out_g)
{
    const int p = blockIdx.x * blockDim.x + threadIdx.x;
    if (p >= BB * NN) return;

    float uv[UU];
    #pragma unroll
    for (int k = 0; k < UU; ++k) uv[k] = u[(size_t)p * UU + k];

    float h1[H1];
    #pragma unroll
    for (int h = 0; h < H1; ++h) {
        float t = b_r1[h];
        const float* w = w_r1 + h * UU;
        #pragma unroll
        for (int k = 0; k < UU; ++k) t = fmaf(uv[k], w[k], t);
        h1[h] = t > 0.f ? t : 0.f;
    }
    float h2[H2];
    #pragma unroll
    for (int o = 0; o < H2; ++o) {
        float t = b_r2[o];
        const float* w = w_r2 + o * H1;
        #pragma unroll
        for (int h = 0; h < H1; ++h) t = fmaf(h1[h], w[h], t);
        h2[o] = t > 0.f ? t : 0.f;
    }
    float lg[OUTC];
    float mx = -3.4e38f;
    #pragma unroll
    for (int o = 0; o < OUTC; ++o) {
        float t = b_r3[o];
        const float* w = w_r3 + o * H2;
        #pragma unroll
        for (int k = 0; k < H2; ++k) t = fmaf(h2[k], w[k], t);
        lg[o] = t;
        mx = fmaxf(mx, t);
    }
    float sum = 0.f;
    #pragma unroll
    for (int o = 0; o < OUTC; ++o) { lg[o] = __expf(lg[o] - mx); sum += lg[o]; }
    const float inv = 1.f / sum;
    #pragma unroll
    for (int o = 0; o < OUTC; ++o) out_probs[(size_t)p * OUTC + o] = lg[o] * inv;

    #pragma unroll
    for (int k = 0; k < UU; ++k) out_u[(size_t)p * UU + k] = uv[k];
    #pragma unroll
    for (int h = 0; h < H1; ++h) out_g[(size_t)p * H1 + h] = g[(size_t)p * H1 + h];
}

// ---------------------------------------------------------------------------
// Host launcher
// ---------------------------------------------------------------------------
extern "C" void kernel_launch(void* const* d_in, const int* in_sizes, int n_in,
                              void* d_out, int out_size, void* d_ws, size_t ws_size,
                              hipStream_t stream)
{
    const float* y      = (const float*)d_in[0];
    const float* A      = (const float*)d_in[1];
    const float* sigma2 = (const float*)d_in[2];
    const float* r      = (const float*)d_in[3];
    const float* Sig    = (const float*)d_in[4];
    const float* w_init = (const float*)d_in[5];
    const float* b_init = (const float*)d_in[6];
    const float* w_p1   = (const float*)d_in[7];
    const float* b_p1   = (const float*)d_in[8];
    const float* w_p2   = (const float*)d_in[9];
    const float* b_p2   = (const float*)d_in[10];
    const float* w_p3   = (const float*)d_in[11];
    const float* b_p3   = (const float*)d_in[12];
    const float* w_ih   = (const float*)d_in[13];
    const float* b_ih   = (const float*)d_in[14];
    const float* w_hh   = (const float*)d_in[15];
    const float* b_hh   = (const float*)d_in[16];
    const float* w_up   = (const float*)d_in[17];
    const float* b_up   = (const float*)d_in[18];
    const float* w_r1   = (const float*)d_in[19];
    const float* b_r1   = (const float*)d_in[20];
    const float* w_r2   = (const float*)d_in[21];
    const float* b_r2   = (const float*)d_in[22];
    const float* w_r3   = (const float*)d_in[23];
    const float* b_r3   = (const float*)d_in[24];

    // Workspace layout (floats)
    float* ws = (float*)d_ws;
    float* G     = ws;                                   // 16*512*512
    float* u     = G    + (size_t)BB * NN * NN;          // 16*512*8
    float* g     = u    + (size_t)BB * NN * UU;          // 16*512*16
    float* Pn    = g    + (size_t)BB * NN * H1;          // 16*512*16
    float* Pj    = Pn   + (size_t)BB * NN * H1;          // 16*512*16
    float* m_sum = Pj   + (size_t)BB * NN * H1;          // 16*512*8

    // Output layout: probs | u | g  (reference return order, flat)
    float* out_probs = (float*)d_out;
    float* out_u     = out_probs + (size_t)BB * NN * OUTC;
    float* out_g     = out_u     + (size_t)BB * NN * UU;

    const int rows = BB * NN;                 // 8192 node rows
    const dim3 rowGrid((rows + 127) / 128);
    const dim3 rowBlk(128);

    // 1) Gram via WMMA: one wave32 per 32x32 block (2x2 WMMA tiles).
    gram_kernel<<<dim3(NN / 32, NN / 32, BB), dim3(32), 0, stream>>>(A, G);

    // 2) u0
    init_kernel<<<rowGrid, rowBlk, 0, stream>>>(A, y, sigma2, G, w_init, b_init, u);

    // 3) L message-passing layers
    for (int l = 0; l < LL; ++l) {
        proj_kernel<<<rowGrid, rowBlk, 0, stream>>>(u, sigma2, w_p1, b_p1, Pn, Pj);
        pair_msg_kernel<<<dim3(rows), dim3(256), 0, stream>>>(
            G, Pn, Pj, w_p1, w_p2, b_p2, w_p3, b_p3, m_sum);
        gru_kernel<<<rowGrid, rowBlk, 0, stream>>>(
            m_sum, r, Sig, w_ih, b_ih, w_hh, b_hh, w_up, b_up, g, u, l == 0 ? 1 : 0);
    }

    // 4) readout + emit (probs, u, g)
    readout_kernel<<<rowGrid, rowBlk, 0, stream>>>(
        u, g, w_r1, b_r1, w_r2, b_r2, w_r3, b_r3, out_probs, out_u, out_g);
}
